// Basic_RSNN_eprop_analog_forward_88845693485086
// MI455X (gfx1250) — matrix operand order repaired and checked by reading, not compiled, not run on hardware
//
#include <hip/hip_runtime.h>
#include <hip/hip_bf16.h>

// ---------------------------------------------------------------------------
// RSNN e-prop (analog tiles) on MI455X / gfx1250.
// f16 WMMA (v_wmma_f32_16x16x32_f16) for the three big GEMMs, f32 accumulate.
// B operands stored pre-swizzled in WMMA fragment order -> one 32B v16h load
// per fragment. 32x32 output tiles per wave (4 WMMAs per 4 fragment loads).
// All LIF state updates fused into GEMM epilogues; spikes double-buffered.
// ---------------------------------------------------------------------------

typedef _Float16 half_t;
typedef __attribute__((ext_vector_type(16))) _Float16 v16h;
typedef __attribute__((ext_vector_type(8)))  _Float16 v8h;
typedef __attribute__((ext_vector_type(8)))  float    v8f;

#define N_IN   700
#define N_INP  704      // padded to multiple of 32 (K dim)
#define N_HID  512
#define N_OUT  20
#define BB     256
#define TT     100
#define TAU    0.6f
#define TAU_O  0.6f
#define THR    0.6f
#define GAMMA  0.3f

// ---------------- WMMA fragment helpers (CDNA5 wave32 layouts) --------------

// A-matrix 16x32 f16, row-major src[M][K]; lane L(0..15): M=L, K {0..15};
// lane L+16: M=L, K {16..31}.  Two aligned v8h loads, no repacking.
__device__ __forceinline__ v16h frag_a(const half_t* __restrict__ src, int lda,
                                       int m0, int k0, int lane) {
  const half_t* row = src + (size_t)(m0 + (lane & 15)) * lda + k0
                          + ((lane & 16) ? 16 : 0);
  v8h lo = *(const v8h*)(row);
  v8h hi = *(const v8h*)(row + 8);
  return __builtin_shufflevector(lo, hi, 0, 1, 2, 3, 4, 5, 6, 7,
                                 8, 9, 10, 11, 12, 13, 14, 15);
}

// Swizzled-B element address: B[k][n] (K x N, N = Nt*16) stored so that the
// WMMA B fragment for tile (k0,n0) is one contiguous 16-half run per lane:
//   lane = (n&15) | (k&16), element = k&15, tile = (k>>5)*Nt + (n>>4).
__device__ __forceinline__ size_t bswz(int k, int n, int Nt) {
  return ((size_t)((k >> 5) * Nt + (n >> 4)) * 32 + ((n & 15) | (k & 16))) * 16
         + (k & 15);
}

// One aligned 32-byte load = entire 32x16 B fragment for this lane.
__device__ __forceinline__ v16h frag_b(const half_t* __restrict__ src, int Nt,
                                       int k0, int n0, int lane) {
  return *(const v16h*)(src +
      ((size_t)((k0 >> 5) * Nt + (n0 >> 4)) * 32 + lane) * 16);
}

__device__ __forceinline__ v8f wmma_f32(v16h a, v16h b, v8f c) {
  return __builtin_amdgcn_wmma_f32_16x16x32_f16(false, a, false, b,
                                                (short)0, c, false, false);
}

// ---------------- prep kernels ---------------------------------------------

__global__ void k_zero(float* __restrict__ p, size_t n) {
  size_t i = (size_t)blockIdx.x * blockDim.x + threadIdx.x;
  size_t stride = (size_t)gridDim.x * blockDim.x;
  for (; i < n; i += stride) p[i] = 0.0f;
}

// W[R][K] f32 row-major -> swizzled f16 B matrix with B[k][n] = W[n][k],
// zero-padded K -> Kpad.  (R = N dimension of the GEMM, Nt = R/16.)
__global__ void k_pack_w(const float* __restrict__ W, half_t* __restrict__ Bsw,
                         int R, int K, int Kpad) {
  int idx = blockIdx.x * blockDim.x + threadIdx.x;
  if (idx >= Kpad * R) return;
  int k = idx / R, n = idx % R;
  float v = (k < K) ? W[(size_t)n * K + k] : 0.0f;
  Bsw[bswz(k, n, R >> 4)] = (half_t)v;
}

// x [B][T][700] f32 -> xf16 [T][B][704] f16 (zero pad), row-major (A operand)
__global__ void k_convert_x(const float* __restrict__ x, half_t* __restrict__ xf16) {
  size_t idx = (size_t)blockIdx.x * blockDim.x + threadIdx.x;
  if (idx >= (size_t)TT * BB * N_INP) return;
  int i = (int)(idx % N_INP);
  size_t rem = idx / N_INP;
  int b = (int)(rem % BB);
  int t = (int)(rem / BB);
  float v = (i < N_IN) ? x[((size_t)b * TT + t) * N_IN + i] : 0.0f;
  xf16[idx] = (half_t)v;
}

// ---------------- per-step kernels -----------------------------------------

__device__ __forceinline__ void hidden_epi(
    v8f c, int mb, int nb, int lane,
    const float* __restrict__ hs32_cur, float* __restrict__ hm,
    float* __restrict__ hs32_nxt, half_t* __restrict__ hs16_nxt,
    float* __restrict__ trec32, half_t* __restrict__ trec16sw,
    float* __restrict__ tout, float* __restrict__ ht) {
  const int n = nb + (lane & 15);
  const int mofs = (lane & 16) ? 8 : 0;
#pragma unroll
  for (int j = 0; j < 8; ++j) {
    const int m = mb + j + mofs;
    const size_t idx = (size_t)m * N_HID + n;
    const float hm_old = hm[idx];
    const float hs_old = hs32_cur[idx];
    const float hmn = TAU * hm_old * (1.0f - hs_old) + c[j];
    const float s   = (hmn >= THR) ? 1.0f : 0.0f;
    const float dv  = fabsf((hmn - THR) * (1.0f / THR));
    const float htv = GAMMA * fmaxf(0.0f, 1.0f - dv);
    hm[idx]       = hmn;
    hs32_nxt[idx] = s;
    hs16_nxt[idx] = (half_t)s;
    ht[idx]       = htv;
    const float tr = TAU * trec32[idx] + s;
    trec32[idx] = tr;
    trec16sw[bswz(m, n, N_HID >> 4)] = (half_t)tr;  // B operand of gr GEMM
    tout[idx]   = TAU_O * tout[idx] + s;
  }
}

// Hidden GEMM + fused LIF / surrogate / trec / tout.
// 32x32 tiles: 8x16 = 128 waves -> 16 blocks of 256 threads.
__global__ void k_hidden(const half_t* __restrict__ xf16t,     // [BB][N_INP]
                         const half_t* __restrict__ wfc1sw,    // swz [704][512]
                         const half_t* __restrict__ wrecsw,    // swz [512][512]
                         const half_t* __restrict__ hs16_cur,  // [BB][N_HID]
                         const float*  __restrict__ hs32_cur,
                         float* __restrict__ hm,
                         float* __restrict__ hs32_nxt, half_t* __restrict__ hs16_nxt,
                         float* __restrict__ trec32,  half_t* __restrict__ trec16sw,
                         float* __restrict__ tout,    float* __restrict__ ht) {
  const int lane = threadIdx.x & 31;
  const int wave = threadIdx.x >> 5;
  const int tile = blockIdx.x * 8 + wave;       // 0..127
  const int m0 = (tile >> 4) * 32;              // batch rows (8 tiles)
  const int n0 = (tile & 15) * 32;              // hidden cols (16 tiles)
  v8f c00 = {}, c01 = {}, c10 = {}, c11 = {};
  for (int k0 = 0; k0 < N_INP; k0 += 32) {
    v16h a0 = frag_a(xf16t, N_INP, m0,      k0, lane);
    v16h a1 = frag_a(xf16t, N_INP, m0 + 16, k0, lane);
    v16h b0 = frag_b(wfc1sw, N_HID >> 4, k0, n0,      lane);
    v16h b1 = frag_b(wfc1sw, N_HID >> 4, k0, n0 + 16, lane);
    c00 = wmma_f32(a0, b0, c00);
    c01 = wmma_f32(a0, b1, c01);
    c10 = wmma_f32(a1, b0, c10);
    c11 = wmma_f32(a1, b1, c11);
  }
  for (int k0 = 0; k0 < N_HID; k0 += 32) {
    v16h a0 = frag_a(hs16_cur, N_HID, m0,      k0, lane);
    v16h a1 = frag_a(hs16_cur, N_HID, m0 + 16, k0, lane);
    v16h b0 = frag_b(wrecsw, N_HID >> 4, k0, n0,      lane);
    v16h b1 = frag_b(wrecsw, N_HID >> 4, k0, n0 + 16, lane);
    c00 = wmma_f32(a0, b0, c00);
    c01 = wmma_f32(a0, b1, c01);
    c10 = wmma_f32(a1, b0, c10);
    c11 = wmma_f32(a1, b1, c11);
  }
  hidden_epi(c00, m0,      n0,      lane, hs32_cur, hm, hs32_nxt, hs16_nxt,
             trec32, trec16sw, tout, ht);
  hidden_epi(c01, m0,      n0 + 16, lane, hs32_cur, hm, hs32_nxt, hs16_nxt,
             trec32, trec16sw, tout, ht);
  hidden_epi(c10, m0 + 16, n0,      lane, hs32_cur, hm, hs32_nxt, hs16_nxt,
             trec32, trec16sw, tout, ht);
  hidden_epi(c11, m0 + 16, n0 + 16, lane, hs32_cur, hm, hs32_nxt, hs16_nxt,
             trec32, trec16sw, tout, ht);
}

// tin = TAU*tin + xt over [BB][N_INP]; f32 linear + swizzled f16 (B of gf GEMM).
__global__ void k_tin(const float* __restrict__ x, float* __restrict__ tin32,
                      half_t* __restrict__ tin16sw, int t) {
  int idx = blockIdx.x * blockDim.x + threadIdx.x;
  if (idx >= BB * N_INP) return;
  int b = idx / N_INP, i = idx % N_INP;
  float xv = (i < N_IN) ? x[((size_t)b * TT + t) * N_IN + i] : 0.0f;
  float tv = TAU * tin32[idx] + xv;
  tin32[idx] = tv;
  tin16sw[bswz(b, i, N_INP >> 4)] = (half_t)tv;   // k = batch, n = input idx
}

// Output LIF: om/os/err + write spikes to d_out. 5120 threads, K=512 dot.
__global__ void k_output(const float* __restrict__ hs32, const float* __restrict__ Wout,
                         const float* __restrict__ label, float* __restrict__ om,
                         float* __restrict__ os, float* __restrict__ err,
                         float* __restrict__ outs, int t) {
  int idx = blockIdx.x * blockDim.x + threadIdx.x;
  if (idx >= BB * N_OUT) return;
  int b = idx / N_OUT, o = idx % N_OUT;
  const float* hrow = hs32 + (size_t)b * N_HID;
  const float* wrow = Wout + (size_t)o * N_HID;
  float acc = 0.0f;
#pragma unroll 4
  for (int h = 0; h < N_HID; ++h) acc = fmaf(hrow[h], wrow[h], acc);
  float omv = TAU * om[idx] * (1.0f - os[idx]) + acc;
  float osv = (omv >= THR) ? 1.0f : 0.0f;
  om[idx] = omv;
  os[idx] = osv;
  size_t lp = ((size_t)b * TT + t) * N_OUT + o;
  err[idx]  = osv - label[lp];
  outs[lp]  = osv;
}

// L = (err @ W_out) * h_t ; store transposed f16 [N_HID][BB] (A of grad GEMMs).
__global__ void k_learnsig(const float* __restrict__ err, const float* __restrict__ Wout,
                           const float* __restrict__ ht, half_t* __restrict__ Lt16) {
  int idx = blockIdx.x * blockDim.x + threadIdx.x;
  if (idx >= BB * N_HID) return;
  int b = idx >> 9, r = idx & (N_HID - 1);
  float acc = 0.0f;
#pragma unroll
  for (int o = 0; o < N_OUT; ++o)
    acc = fmaf(err[b * N_OUT + o], Wout[(size_t)o * N_HID + r], acc);
  Lt16[(size_t)r * BB + b] = (half_t)(acc * ht[idx]);
}

__device__ __forceinline__ void grad_epi(v8f c, int mb, int nb, int lane,
                                         float* __restrict__ G, int Nreal, int ldg) {
  const int n = nb + (lane & 15);
  if (n >= Nreal) return;
  const int mofs = (lane & 16) ? 8 : 0;
#pragma unroll
  for (int j = 0; j < 8; ++j) {
    const int m = mb + j + mofs;
    G[(size_t)m * ldg + n] += 0.1f * c[j];
  }
}

// G[512][Nreal] += 0.1 * Lt[512][256] @ Bsw[256][Npad].  32x32 tiles, K=256.
__global__ void k_grad(const half_t* __restrict__ At, const half_t* __restrict__ Bsw,
                       float* __restrict__ G, int Npad, int Nreal, int ldg) {
  const int lane = threadIdx.x & 31;
  const int wave = threadIdx.x >> 5;
  const int nt32 = Npad >> 5;
  const int tile = blockIdx.x * 8 + wave;
  const int m0 = (tile / nt32) * 32;
  const int n0 = (tile % nt32) * 32;
  v8f c00 = {}, c01 = {}, c10 = {}, c11 = {};
#pragma unroll
  for (int k0 = 0; k0 < BB; k0 += 32) {
    v16h a0 = frag_a(At, BB, m0,      k0, lane);
    v16h a1 = frag_a(At, BB, m0 + 16, k0, lane);
    v16h b0 = frag_b(Bsw, Npad >> 4, k0, n0,      lane);
    v16h b1 = frag_b(Bsw, Npad >> 4, k0, n0 + 16, lane);
    c00 = wmma_f32(a0, b0, c00);
    c01 = wmma_f32(a0, b1, c01);
    c10 = wmma_f32(a1, b0, c10);
    c11 = wmma_f32(a1, b1, c11);
  }
  grad_epi(c00, m0,      n0,      lane, G, Nreal, ldg);
  grad_epi(c01, m0,      n0 + 16, lane, G, Nreal, ldg);
  grad_epi(c10, m0 + 16, n0,      lane, G, Nreal, ldg);
  grad_epi(c11, m0 + 16, n0 + 16, lane, G, Nreal, ldg);
}

// go[20][512] += 0.1 * err^T @ tout   (tiny: 10240 outputs x K=256)
__global__ void k_go(const float* __restrict__ err, const float* __restrict__ tout,
                     float* __restrict__ go) {
  int idx = blockIdx.x * blockDim.x + threadIdx.x;
  if (idx >= N_OUT * N_HID) return;
  int o = idx / N_HID, r = idx % N_HID;
  float acc = 0.0f;
#pragma unroll 4
  for (int b = 0; b < BB; ++b)
    acc = fmaf(err[b * N_OUT + o], tout[(size_t)b * N_HID + r], acc);
  go[idx] += 0.1f * acc;
}

// ---------------------------------------------------------------------------

extern "C" void kernel_launch(void* const* d_in, const int* in_sizes, int n_in,
                              void* d_out, int out_size, void* d_ws, size_t ws_size,
                              hipStream_t stream) {
  const float* x     = (const float*)d_in[0];  // [256][100][700]
  const float* label = (const float*)d_in[1];  // [256][100][20]
  const float* Wfc1  = (const float*)d_in[2];  // [512][700]
  const float* Wrec  = (const float*)d_in[3];  // [512][512]
  const float* Wout  = (const float*)d_in[4];  // [20][512]
  float* out = (float*)d_out;

  // d_out layout: outs[256*100*20] | gf[512*700] | gr[512*512] | go[20*512]
  float* outs = out;
  float* gf   = out + (size_t)BB * TT * N_OUT;
  float* gr   = gf  + (size_t)N_HID * N_IN;
  float* go   = gr  + (size_t)N_HID * N_HID;
  const size_t gradFloats = (size_t)N_HID * N_IN + (size_t)N_HID * N_HID
                          + (size_t)N_OUT * N_HID;

  // ---- workspace carve-out (256B aligned slices) ----
  char* w = (char*)d_ws;
  size_t off = 0;
  auto take = [&](size_t bytes) -> char* {
    char* p = w + off;
    off = (off + bytes + 255) & ~(size_t)255;
    return p;
  };
  half_t* xf16    = (half_t*)take((size_t)TT * BB * N_INP * 2);   // 36.0 MB
  half_t* wfc1sw  = (half_t*)take((size_t)N_INP * N_HID * 2);
  half_t* wrecsw  = (half_t*)take((size_t)N_HID * N_HID * 2);
  size_t stateStart = off;
  float*  hm      = (float*) take((size_t)BB * N_HID * 4);
  float*  hs32[2] = { (float*)take((size_t)BB * N_HID * 4),
                      (float*)take((size_t)BB * N_HID * 4) };
  half_t* hs16[2] = { (half_t*)take((size_t)BB * N_HID * 2),
                      (half_t*)take((size_t)BB * N_HID * 2) };
  float*  om      = (float*) take((size_t)BB * N_OUT * 4);
  float*  osb     = (float*) take((size_t)BB * N_OUT * 4);
  float*  err     = (float*) take((size_t)BB * N_OUT * 4);
  float*  tin32   = (float*) take((size_t)BB * N_INP * 4);
  half_t* tin16sw = (half_t*)take((size_t)BB * N_INP * 2);
  float*  trec32  = (float*) take((size_t)BB * N_HID * 4);
  half_t* trec16sw= (half_t*)take((size_t)BB * N_HID * 2);
  float*  tout    = (float*) take((size_t)BB * N_HID * 4);
  float*  ht      = (float*) take((size_t)BB * N_HID * 4);
  half_t* Lt16    = (half_t*)take((size_t)N_HID * BB * 2);
  size_t stateFloats = (off - stateStart) / 4;

  // ---- one-time prep (stream-ordered; deterministic every call) ----
  k_zero<<<512, 256, 0, stream>>>((float*)(w + stateStart), stateFloats);
  k_zero<<<512, 256, 0, stream>>>(gf, gradFloats);
  {
    size_t n = (size_t)TT * BB * N_INP;
    k_convert_x<<<(int)((n + 255) / 256), 256, 0, stream>>>(x, xf16);
  }
  k_pack_w<<<(N_INP * N_HID + 255) / 256, 256, 0, stream>>>(
      Wfc1, wfc1sw, N_HID, N_IN, N_INP);
  k_pack_w<<<(N_HID * N_HID + 255) / 256, 256, 0, stream>>>(
      Wrec, wrecsw, N_HID, N_HID, N_HID);

  // ---- 100-step scan: 7 stream-ordered kernels per step ----
  for (int t = 0; t < TT; ++t) {
    const int cur = t & 1, nxt = cur ^ 1;
    // 128 waves of 32x32 tiles -> 16 blocks
    k_hidden<<<16, 256, 0, stream>>>(
        xf16 + (size_t)t * BB * N_INP, wfc1sw, wrecsw,
        hs16[cur], hs32[cur],
        hm, hs32[nxt], hs16[nxt], trec32, trec16sw, tout, ht);
    k_tin<<<(BB * N_INP + 255) / 256, 256, 0, stream>>>(x, tin32, tin16sw, t);
    k_output<<<(BB * N_OUT + 255) / 256, 256, 0, stream>>>(
        hs32[nxt], Wout, label, om, osb, err, outs, t);
    k_learnsig<<<(BB * N_HID + 255) / 256, 256, 0, stream>>>(err, Wout, ht, Lt16);
    // gf: M=512, N=704 (700 real): 16x22 = 352 tiles -> 44 blocks
    k_grad<<<44, 256, 0, stream>>>(Lt16, tin16sw, gf, N_INP, N_IN, N_IN);
    // gr: M=512, N=512: 16x16 = 256 tiles -> 32 blocks
    k_grad<<<32, 256, 0, stream>>>(Lt16, trec16sw, gr, N_HID, N_HID, N_HID);
    k_go<<<(N_OUT * N_HID + 255) / 256, 256, 0, stream>>>(err, tout, go);
  }
}
